// BinaryClassifier_8675833937973
// MI455X (gfx1250) — matrix operand order repaired and checked by reading, not compiled, not run on hardware
//
#include <hip/hip_runtime.h>
#include <cstdint>
#include <cstddef>

// B=2, L=512, D_IN=400 (pad 416), D=256, H=8, DK=32, D_FF=1024, N_LAYERS=2, NCLS=2

typedef __attribute__((ext_vector_type(16))) __bf16 bf16x16;
typedef __attribute__((ext_vector_type(8)))  __bf16 bf16x8;
typedef __attribute__((ext_vector_type(8)))  float  floatx8;
typedef __attribute__((ext_vector_type(4)))  float  floatx4;

#if defined(__has_builtin)
#if __has_builtin(__builtin_amdgcn_global_load_async_to_lds_b128) && \
    __has_builtin(__builtin_amdgcn_s_wait_asynccnt)
#define USE_ASYNC_LDS 1
#endif
#endif

static __device__ __forceinline__ bf16x16 cat8(bf16x8 lo, bf16x8 hi) {
  return __builtin_shufflevector(lo, hi, 0, 1, 2, 3, 4, 5, 6, 7, 8, 9, 10, 11, 12, 13, 14, 15);
}
// A fragment (16x32 bf16): row base Arow, K pairs split by lane-half. Two b128 loads.
static __device__ __forceinline__ bf16x16 load_a_frag(const __bf16* Arow, int k0, int half) {
  bf16x8 lo = *(const bf16x8*)(Arow + k0 + half * 8);
  bf16x8 hi = *(const bf16x8*)(Arow + k0 + 16 + half * 8);
  return cat8(lo, hi);
}
// B fragment (32x16 bf16) from transposed weight row (col-major => contiguous K). Two b128 loads.
static __device__ __forceinline__ bf16x16 load_b_frag(const __bf16* Wrow, int k0, int half) {
  bf16x8 lo = *(const bf16x8*)(Wrow + k0 + half * 8);
  bf16x8 hi = *(const bf16x8*)(Wrow + k0 + 16 + half * 8);
  return cat8(lo, hi);
}

// ---------------------------------------------------------------------------
// Activation f32 -> bf16 with K-padding: out[M,Kp], zero beyond K
// ---------------------------------------------------------------------------
__global__ void cvt_pad_kernel(const float* __restrict__ in, __bf16* __restrict__ out,
                               int K, int Kp, int M) {
  int idx = blockIdx.x * blockDim.x + threadIdx.x;
  if (idx >= M * Kp) return;
  int row = idx / Kp, k = idx - row * Kp;
  out[idx] = (k < K) ? (__bf16)in[(size_t)row * K + k] : (__bf16)0.0f;
}

// ---------------------------------------------------------------------------
// Weight W[K,N] f32 -> Wt[N,Kp] bf16 (transpose + pad): B-frags become b128 loads
// ---------------------------------------------------------------------------
__global__ void wt_cvt_kernel(const float* __restrict__ W, __bf16* __restrict__ Wt,
                              int K, int Kp, int N) {
  int idx = blockIdx.x * blockDim.x + threadIdx.x;
  if (idx >= N * Kp) return;
  int n = idx / Kp, k = idx - n * Kp;
  Wt[idx] = (k < K) ? (__bf16)W[(size_t)k * N + n] : (__bf16)0.0f;
}

// ---------------------------------------------------------------------------
// V (B*L, 256) bf16 -> Vt (B,H,DK,L) bf16 so attn P@V B-frags are contiguous
// ---------------------------------------------------------------------------
__global__ void vtr_kernel(const __bf16* __restrict__ vb, __bf16* __restrict__ vt) {
  int idx = blockIdx.x * blockDim.x + threadIdx.x;
  if (idx >= 1024 * 256) return;
  int c = idx & 255, row = idx >> 8;  // row = b*512 + i
  int b = row >> 9, i = row & 511;
  int h = c >> 5, dk = c & 31;
  vt[(((size_t)(b * 8 + h) * 32) + dk) * 512 + i] = vb[idx];
}

// ---------------------------------------------------------------------------
// pos_w head-mean reduction: pwred[d*8+h] = mean_dk pos_w[d, h*32+dk]
// ---------------------------------------------------------------------------
__global__ void pwred_kernel(const float* __restrict__ pw, const float* __restrict__ pb,
                             float* __restrict__ pwred, float* __restrict__ pbred) {
  int idx = blockIdx.x * blockDim.x + threadIdx.x;
  if (idx < 2048) {
    int d = idx >> 3, h = idx & 7;
    float s = 0.0f;
    for (int t = 0; t < 32; ++t) s += pw[d * 256 + h * 32 + t];
    pwred[idx] = s * (1.0f / 32.0f);
  }
  if (idx < 8) {
    float s = 0.0f;
    for (int t = 0; t < 32; ++t) s += pb[idx * 32 + t];
    pbred[idx] = s * (1.0f / 32.0f);
  }
}

// ---------------------------------------------------------------------------
// Positional bias + mask: biasb[(b,h,i,j)] = masked ? -1e9 : pos_bias
// pwred staged in LDS via async global->LDS DMA when available (ASYNCcnt).
// ---------------------------------------------------------------------------
__global__ void posbias_kernel(const float* __restrict__ pos_ind,
                               const float* __restrict__ fmask,
                               const float* __restrict__ pwred,
                               const float* __restrict__ pbred,
                               float* __restrict__ biasb) {
  __shared__ float wsh[2048];
  __shared__ float bsh[8];
#ifdef USE_ASYNC_LDS
  {
    // toolchain signature (from diagnostics): param0 = AS1 int4*, so cast to match
    typedef int v4i_ __attribute__((vector_size(16)));
    typedef __attribute__((address_space(1))) v4i_* gp_t;
    typedef __attribute__((address_space(3))) v4i_* lp_t;
    gp_t g = (gp_t)(unsigned long long)(const char*)pwred;
    lp_t l = (lp_t)wsh;
    for (int t = threadIdx.x; t < 512; t += blockDim.x)  // 512 x 16B = 8KB
      __builtin_amdgcn_global_load_async_to_lds_b128(g + t, l + t, 0, 0);
    if (threadIdx.x < 8) bsh[threadIdx.x] = pbred[threadIdx.x];
    __builtin_amdgcn_s_wait_asynccnt(0);
  }
#else
  for (int t = threadIdx.x; t < 2048; t += blockDim.x) wsh[t] = pwred[t];
  if (threadIdx.x < 8) bsh[threadIdx.x] = pbred[threadIdx.x];
#endif
  __syncthreads();

  int idx = blockIdx.x * blockDim.x + threadIdx.x;
  if (idx >= 2 * 512 * 512) return;
  int j = idx & 511, i = (idx >> 9) & 511, b = idx >> 18;

  float p0 = floorf(pos_ind[(size_t)idx * 2 + 0]);
  float p1 = floorf(pos_ind[(size_t)idx * 2 + 1]);

  float acc[8];
#pragma unroll
  for (int h = 0; h < 8; ++h) acc[h] = bsh[h];

  const float invr = __expf(-0.14391157f);  // 10000^(-1/64)
  float invf = 1.0f;
  for (int f = 0; f < 64; ++f) {
    float d0 = p0 * invf, d1 = p1 * invf;
    float s0 = __sinf(d0), c0 = __cosf(d0);
    float s1 = __sinf(d1), c1 = __cosf(d1);
    const float* w0 = wsh + f * 8;
#pragma unroll
    for (int h = 0; h < 8; ++h)
      acc[h] += s0 * w0[h] + c0 * w0[512 + h] + s1 * w0[1024 + h] + c1 * w0[1536 + h];
    invf *= invr;
  }

  bool pad_i = fmask[b * 512 + i] == 0.0f;
  bool pad_j = fmask[b * 512 + j] == 0.0f;
  int di = i - j;
  bool local_np = (di >= 9) || (di <= -8);
  int ad = di < 0 ? -di : di;
  bool dil_np = (ad & 15) != 0;
  bool masked = (dil_np || pad_i || pad_j) && (local_np || pad_j);

#pragma unroll
  for (int h = 0; h < 8; ++h) {
    size_t o = (((size_t)(b * 8 + h) * 512) + i) * 512 + j;
    biasb[o] = masked ? -1.0e9f : acc[h];
  }
}

// ---------------------------------------------------------------------------
// WMMA GEMM: C[M,N] = act(A[M,K](bf16) @ Wt[N,K](bf16, transposed) + bias)
// One wave per 16x16 tile; K multiple of 32; all frag loads are b128.
// ACT: 0=none, 1=SELU, 2=ReLU. outF/outB each optional.
// ---------------------------------------------------------------------------
template <int ACT>
__global__ void gemm_wmma_kernel(const __bf16* __restrict__ A, const __bf16* __restrict__ Wt,
                                 const float* __restrict__ bias, float* __restrict__ outF,
                                 __bf16* __restrict__ outB, int M, int N, int K) {
  int lane = threadIdx.x & 31;
  int wave = blockIdx.x * (blockDim.x >> 5) + (threadIdx.x >> 5);
  int tilesN = N >> 4;
  int tilesM = M >> 4;
  if (wave >= tilesM * tilesN) return;  // uniform whole-wave exit: EXEC all ones for WMMA
  int tm = wave / tilesN, tn = wave % tilesN;
  int row0 = tm << 4, col0 = tn << 4;
  int r16 = lane & 15, half = lane >> 4;

  const __bf16* Arow = A + (size_t)(row0 + r16) * K;
  const __bf16* Wrow = Wt + (size_t)(col0 + r16) * K;

  floatx8 c = {};
  for (int k0 = 0; k0 < K; k0 += 32) {
    bf16x16 a = load_a_frag(Arow, k0, half);
    bf16x16 b = load_b_frag(Wrow, k0, half);
    c = __builtin_amdgcn_wmma_f32_16x16x32_bf16(false, a, false, b, (short)0, c, false, false);
  }

  float bval = bias ? bias[col0 + r16] : 0.0f;
#pragma unroll
  for (int r = 0; r < 8; ++r) {
    int m = r + half * 8;
    float y = c[r] + bval;
    if (ACT == 1) {  // SELU
      const float sc = 1.0507009873554805f, al = 1.6732632423543772f;
      y = sc * (y > 0.0f ? y : al * (__expf(y) - 1.0f));
    } else if (ACT == 2) {  // ReLU
      y = fmaxf(y, 0.0f);
    }
    size_t oi = (size_t)(row0 + m) * N + col0 + r16;
    if (outF) outF[oi] = y;
    if (outB) outB[oi] = (__bf16)y;
  }
}

// ---------------------------------------------------------------------------
// Attention: one wave per (b, h, 16-row q tile).
//  pass1: scores(16x512) = (q kT)*scale + bias -> LDS (WMMA, K=DK=32)
//  softmax in LDS; pass2: o(16x32) = P @ Vt     (WMMA over K=512)
// ---------------------------------------------------------------------------
__global__ void attn_kernel(const __bf16* __restrict__ qb, const __bf16* __restrict__ kb,
                            const __bf16* __restrict__ vt, const float* __restrict__ biasb,
                            __bf16* __restrict__ ob) {
  __shared__ float s[16 * 512];  // 32 KB score tile
  int lane = threadIdx.x;        // blockDim == 32
  int t = blockIdx.x;            // 0..511
  int tile = t & 31;
  int h = (t >> 5) & 7;
  int b = t >> 8;
  int i0 = tile << 4;
  int r16 = lane & 15, half = lane >> 4;
  const float scale = 0.17677669529663687f;  // 1/sqrt(32)

  // q fragment (A, 16x32): loaded once, reused for all key tiles
  const __bf16* qrow = qb + ((size_t)(b * 512 + i0 + r16) * 256 + h * 32);
  bf16x16 aq = load_a_frag(qrow, 0, half);

  const float* biasrow = biasb + (size_t)((b * 8 + h) * 512) * 512;

  // ---- pass 1: scores ----
  for (int n0 = 0; n0 < 512; n0 += 16) {
    const __bf16* krow = kb + ((size_t)(b * 512 + n0 + r16) * 256 + h * 32);
    bf16x16 bk = load_b_frag(krow, 0, half);  // B = K^T
    floatx8 c = {};
    c = __builtin_amdgcn_wmma_f32_16x16x32_bf16(false, aq, false, bk, (short)0, c, false, false);
#pragma unroll
    for (int r = 0; r < 8; ++r) {
      int m = r + half * 8;
      s[m * 512 + n0 + r16] = c[r] * scale + biasrow[(size_t)(i0 + m) * 512 + n0 + r16];
    }
  }
  __syncthreads();

  // ---- softmax: row = r16, two lanes per row (segments of 256) ----
  {
    float* row = s + r16 * 512;
    int beg = half * 256;
    float mx = -3.0e38f;
    for (int j = 0; j < 256; ++j) mx = fmaxf(mx, row[beg + j]);
    mx = fmaxf(mx, __shfl_xor(mx, 16, 32));
    float sum = 0.0f;
    for (int j = 0; j < 256; ++j) {
      float e = __expf(row[beg + j] - mx);
      row[beg + j] = e;
      sum += e;
    }
    sum += __shfl_xor(sum, 16, 32);
    float inv = 1.0f / sum;
    for (int j = 0; j < 256; ++j) row[beg + j] *= inv;
  }
  __syncthreads();

  // ---- pass 2: o = P @ V (Vt rows are contiguous along keys) ----
  const __bf16* vrow0 = vt + ((size_t)((b * 8 + h) * 32) + r16) * 512;
  const __bf16* vrow1 = vt + ((size_t)((b * 8 + h) * 32) + 16 + r16) * 512;
  const float* srow = s + r16 * 512;
  floatx8 o0 = {}, o1 = {};
  for (int k0 = 0; k0 < 512; k0 += 32) {
    floatx4 f0 = *(const floatx4*)(srow + k0 + half * 8);
    floatx4 f1 = *(const floatx4*)(srow + k0 + half * 8 + 4);
    floatx4 f2 = *(const floatx4*)(srow + k0 + 16 + half * 8);
    floatx4 f3 = *(const floatx4*)(srow + k0 + 16 + half * 8 + 4);
    bf16x16 ap;
#pragma unroll
    for (int j = 0; j < 4; ++j) {
      ap[j] = (__bf16)f0[j];
      ap[4 + j] = (__bf16)f1[j];
      ap[8 + j] = (__bf16)f2[j];
      ap[12 + j] = (__bf16)f3[j];
    }
    bf16x16 bv0 = load_b_frag(vrow0, k0, half);
    bf16x16 bv1 = load_b_frag(vrow1, k0, half);
    o0 = __builtin_amdgcn_wmma_f32_16x16x32_bf16(false, ap, false, bv0, (short)0, o0, false, false);
    o1 = __builtin_amdgcn_wmma_f32_16x16x32_bf16(false, ap, false, bv1, (short)0, o1, false, false);
  }
#pragma unroll
  for (int r = 0; r < 8; ++r) {
    int m = r + half * 8;
    size_t base = (size_t)(b * 512 + i0 + m) * 256 + h * 32;
    ob[base + r16] = (__bf16)o0[r];
    ob[base + 16 + r16] = (__bf16)o1[r];
  }
}

// ---------------------------------------------------------------------------
// Residual add + LayerNorm (D=256), one wave per row. Writes f32 + bf16 x.
// ---------------------------------------------------------------------------
__global__ void ln_kernel(const float* resid, const float* __restrict__ y,
                          const float* __restrict__ g, const float* __restrict__ be,
                          float* xf, __bf16* __restrict__ xb, int M) {
  int lane = threadIdx.x & 31;
  int row = blockIdx.x * (blockDim.x >> 5) + (threadIdx.x >> 5);
  if (row >= M) return;
  const float* rp = resid + (size_t)row * 256;
  const float* yp = y + (size_t)row * 256;
  float v[8];
  float sum = 0.0f;
#pragma unroll
  for (int j = 0; j < 8; ++j) {
    v[j] = rp[lane + 32 * j] + yp[lane + 32 * j];
    sum += v[j];
  }
#pragma unroll
  for (int off = 16; off; off >>= 1) sum += __shfl_xor(sum, off, 32);
  float mean = sum * (1.0f / 256.0f);
  float var = 0.0f;
#pragma unroll
  for (int j = 0; j < 8; ++j) {
    float d = v[j] - mean;
    var += d * d;
  }
#pragma unroll
  for (int off = 16; off; off >>= 1) var += __shfl_xor(var, off, 32);
  var *= (1.0f / 256.0f);
  float rstd = rsqrtf(var + 1e-5f);
#pragma unroll
  for (int j = 0; j < 8; ++j) {
    int cidx = lane + 32 * j;
    float o = (v[j] - mean) * rstd * g[cidx] + be[cidx];
    xf[(size_t)row * 256 + cidx] = o;
    xb[(size_t)row * 256 + cidx] = (__bf16)o;
  }
}

// ---------------------------------------------------------------------------
// Classifier: logits = x @ cls_w + cls_b (256 -> 2), softmax. One thread/row.
// ---------------------------------------------------------------------------
__global__ void cls_kernel(const float* __restrict__ xf, const float* __restrict__ w,
                           const float* __restrict__ bb, float* __restrict__ out, int M) {
  int row = blockIdx.x * blockDim.x + threadIdx.x;
  if (row >= M) return;
  const float* xp = xf + (size_t)row * 256;
  float l0 = bb[0], l1 = bb[1];
  for (int d = 0; d < 256; ++d) {
    float x = xp[d];
    l0 += x * w[d * 2 + 0];
    l1 += x * w[d * 2 + 1];
  }
  float m = fmaxf(l0, l1);
  float e0 = __expf(l0 - m), e1 = __expf(l1 - m);
  float inv = 1.0f / (e0 + e1);
  out[row * 2 + 0] = e0 * inv;
  out[row * 2 + 1] = e1 * inv;
}

// ---------------------------------------------------------------------------
// Launch
// ---------------------------------------------------------------------------
extern "C" void kernel_launch(void* const* d_in, const int* in_sizes, int n_in,
                              void* d_out, int out_size, void* d_ws, size_t ws_size,
                              hipStream_t stream) {
  (void)in_sizes; (void)n_in; (void)out_size; (void)ws_size;
  const float* feature  = (const float*)d_in[0];
  const float* pos_ind  = (const float*)d_in[1];
  const float* fmask    = (const float*)d_in[2];
  const float* w_reduce = (const float*)d_in[3];
  const float* b_reduce = (const float*)d_in[4];
  const float* pos_w    = (const float*)d_in[5];
  const float* pos_b    = (const float*)d_in[6];
  const float* wq       = (const float*)d_in[7];
  const float* wk       = (const float*)d_in[8];
  const float* wv       = (const float*)d_in[9];
  const float* wo       = (const float*)d_in[10];
  const float* ln1_g    = (const float*)d_in[11];
  const float* ln1_b    = (const float*)d_in[12];
  const float* w1       = (const float*)d_in[13];
  const float* b1       = (const float*)d_in[14];
  const float* w2       = (const float*)d_in[15];
  const float* b2       = (const float*)d_in[16];
  const float* ln2_g    = (const float*)d_in[17];
  const float* ln2_b    = (const float*)d_in[18];
  const float* cls_w    = (const float*)d_in[19];
  const float* cls_b    = (const float*)d_in[20];
  float* out = (float*)d_out;

  // workspace carve-up
  char* p = (char*)d_ws;
  auto alloc = [&](size_t sz) -> char* {
    char* r = p;
    p += (sz + 255) & ~(size_t)255;
    return r;
  };
  __bf16* featb = (__bf16*)alloc((size_t)1024 * 416 * 2);   // padded K
  __bf16* xb    = (__bf16*)alloc((size_t)1024 * 256 * 2);
  float*  xf    = (float*) alloc((size_t)1024 * 256 * 4);
  __bf16* qb    = (__bf16*)alloc((size_t)1024 * 256 * 2);
  __bf16* kb    = (__bf16*)alloc((size_t)1024 * 256 * 2);
  __bf16* vb    = (__bf16*)alloc((size_t)1024 * 256 * 2);
  __bf16* vt    = (__bf16*)alloc((size_t)1024 * 256 * 2);
  __bf16* obuf  = (__bf16*)alloc((size_t)1024 * 256 * 2);
  float*  tmpF  = (float*) alloc((size_t)1024 * 256 * 4);
  __bf16* h1b   = (__bf16*)alloc((size_t)1024 * 1024 * 2);
  float*  pwred = (float*) alloc((size_t)2048 * 4);
  float*  pbred = (float*) alloc((size_t)8 * 4);
  float*  biasb = (float*) alloc((size_t)2 * 8 * 512 * 512 * 4);
  // transposed bf16 weights
  __bf16* wrt  = (__bf16*)alloc((size_t)256 * 416 * 2);        // w_reduce^T (padded)
  __bf16* wqt  = (__bf16*)alloc((size_t)2 * 256 * 256 * 2);
  __bf16* wkt  = (__bf16*)alloc((size_t)2 * 256 * 256 * 2);
  __bf16* wvt  = (__bf16*)alloc((size_t)2 * 256 * 256 * 2);
  __bf16* wot  = (__bf16*)alloc((size_t)2 * 256 * 256 * 2);
  __bf16* w1t  = (__bf16*)alloc((size_t)2 * 1024 * 256 * 2);   // N=1024, Kp=256
  __bf16* w2t  = (__bf16*)alloc((size_t)2 * 256 * 1024 * 2);   // N=256,  Kp=1024

  // ---- weight / activation conversion (bf16, transposed, K-padded) ----
  cvt_pad_kernel<<<(1024 * 416 + 255) / 256, 256, 0, stream>>>(feature, featb, 400, 416, 1024);
  wt_cvt_kernel<<<(256 * 416 + 255) / 256, 256, 0, stream>>>(w_reduce, wrt, 400, 416, 256);
  for (int l = 0; l < 2; ++l) {
    wt_cvt_kernel<<<256, 256, 0, stream>>>(wq + (size_t)l * 65536, wqt + (size_t)l * 65536, 256, 256, 256);
    wt_cvt_kernel<<<256, 256, 0, stream>>>(wk + (size_t)l * 65536, wkt + (size_t)l * 65536, 256, 256, 256);
    wt_cvt_kernel<<<256, 256, 0, stream>>>(wv + (size_t)l * 65536, wvt + (size_t)l * 65536, 256, 256, 256);
    wt_cvt_kernel<<<256, 256, 0, stream>>>(wo + (size_t)l * 65536, wot + (size_t)l * 65536, 256, 256, 256);
    wt_cvt_kernel<<<1024, 256, 0, stream>>>(w1 + (size_t)l * 262144, w1t + (size_t)l * 262144, 256, 256, 1024);
    wt_cvt_kernel<<<1024, 256, 0, stream>>>(w2 + (size_t)l * 262144, w2t + (size_t)l * 262144, 1024, 1024, 256);
  }

  // ---- positional bias (+ mask folded) ----
  pwred_kernel<<<8, 256, 0, stream>>>(pos_w, pos_b, pwred, pbred);
  posbias_kernel<<<2048, 256, 0, stream>>>(pos_ind, fmask, pwred, pbred, biasb);

  // ---- reduce layer: x = selu(feature @ w_reduce + b) ----
  gemm_wmma_kernel<1><<<256, 128, 0, stream>>>(featb, wrt, b_reduce, xf, xb, 1024, 256, 416);

  for (int l = 0; l < 2; ++l) {
    const __bf16* wql = wqt + (size_t)l * 65536;
    const __bf16* wkl = wkt + (size_t)l * 65536;
    const __bf16* wvl = wvt + (size_t)l * 65536;
    const __bf16* wol = wot + (size_t)l * 65536;
    // QKV projections
    gemm_wmma_kernel<0><<<256, 128, 0, stream>>>(xb, wql, nullptr, nullptr, qb, 1024, 256, 256);
    gemm_wmma_kernel<0><<<256, 128, 0, stream>>>(xb, wkl, nullptr, nullptr, kb, 1024, 256, 256);
    gemm_wmma_kernel<0><<<256, 128, 0, stream>>>(xb, wvl, nullptr, nullptr, vb, 1024, 256, 256);
    vtr_kernel<<<1024, 256, 0, stream>>>(vb, vt);
    // attention
    attn_kernel<<<512, 32, 0, stream>>>(qb, kb, vt, biasb, obuf);
    // output projection + residual LayerNorm
    gemm_wmma_kernel<0><<<256, 128, 0, stream>>>(obuf, wol, nullptr, tmpF, nullptr, 1024, 256, 256);
    ln_kernel<<<256, 128, 0, stream>>>(xf, tmpF, ln1_g + l * 256, ln1_b + l * 256, xf, xb, 1024);
    // FFN
    gemm_wmma_kernel<2><<<1024, 128, 0, stream>>>(xb, w1t + (size_t)l * 262144, b1 + l * 1024,
                                                  nullptr, h1b, 1024, 1024, 256);
    gemm_wmma_kernel<0><<<256, 128, 0, stream>>>(h1b, w2t + (size_t)l * 262144, b2 + l * 256,
                                                 tmpF, nullptr, 1024, 256, 1024);
    ln_kernel<<<256, 128, 0, stream>>>(xf, tmpF, ln2_g + l * 256, ln2_b + l * 256, xf, xb, 1024);
  }
  // classifier + softmax
  cls_kernel<<<4, 256, 0, stream>>>(xf, cls_w, cls_b, out, 1024);
}
